// VOXCPM_MAIN_403726926221
// MI455X (gfx1250) — compile-verified, matrix operand order checked by reading, not run on hardware
//
#include <hip/hip_runtime.h>
#include <hip/hip_bf16.h>

// =====================================================================
// VoxCPM-style 12-layer transformer forward for MI455X (gfx1250).
// All matmuls via v_wmma_f32_16x16x32_bf16 (bf16 inputs, f32 accum).
// Weights stay fp32 in HBM; converted to bf16 while staging into LDS.
// Attention is flash-style (fused QK^T -> online softmax -> PV), so no
// giant score buffer is materialized. Workspace use: ~58 MB.
// =====================================================================

typedef __attribute__((ext_vector_type(16))) __bf16 v16bf;
typedef __attribute__((ext_vector_type(8)))  float  v8f;
typedef unsigned short ushort_t;

#define HID   1024
#define TQ    1024          // ids_len
#define HIST  256
#define KVSEQ 1280
#define DFF   4096
#define NHEAD 16
#define HD    64
#define QOUT  1024
#define QKVW  1280          // Q_OUT + K_OUT + V_OUT

union FragU { uint4 q[2]; v16bf v; };

__device__ __forceinline__ ushort_t f2b(float f) {
    union { float f; unsigned u; } v; v.f = f;
    unsigned r = (v.u + 0x7FFFu + ((v.u >> 16) & 1u)) >> 16;   // RNE-ish
    return (ushort_t)r;
}

// Load a 16x32 bf16 A/B fragment from LDS (row-major, `ld` ushort stride).
// ISA layout (16-bit A 16x32): lanes 0-15 & 16-31 both hold M=lane&15;
// lane-half selects K {0..7,16..23} vs {8..15,24..31}.
__device__ __forceinline__ v16bf frag_ld(const ushort_t* base, int ld,
                                         int lane, int kbase) {
    int m  = lane & 15;
    int ko = kbase + ((lane >> 4) << 3);
    const ushort_t* p = base + m * ld + ko;
    FragU f;
    f.q[0] = *(const uint4*)p;
    f.q[1] = *(const uint4*)(p + 16);
    return f.v;
}

__device__ __forceinline__ v8f wmma_bf16(v16bf a, v16bf b, v8f c) {
    return __builtin_amdgcn_wmma_f32_16x16x32_bf16(
        /*neg_a=*/false, a, /*neg_b=*/false, b,
        /*c_mod=*/(short)0, c, /*reuse_a=*/false, /*reuse_b=*/false);
}

// ---------------------------------------------------------------------
// Generic NT GEMM: C[M,N] (f32) = A(bf16,[M,K],lda) * B(f32,[N,K],ldb)^T
// (+ residual R if RESID). Block tile 128x128, K-step 32, 8 waves (4x2),
// each wave 32x64 = 8 accumulators -> 8 WMMA per K-step.
// M,N must be multiples of 128 (M=512 ok via grid), K multiple of 32.
// ---------------------------------------------------------------------
template<bool RESID>
__global__ __launch_bounds__(256)
void gemm_nt_f32w(const ushort_t* __restrict__ A, int lda,
                  const float*    __restrict__ B, int ldb,
                  float*          __restrict__ C, int ldc,
                  const float*    __restrict__ R, int ldr,
                  int K)
{
    __shared__ ushort_t sA[128][40];   // +8 pad, rows 16B-aligned
    __shared__ ushort_t sB[128][40];

    const int tid  = threadIdx.x;
    const int lane = tid & 31;
    const int wid  = tid >> 5;
    const int wm   = wid & 3;          // 4 waves along M
    const int wn   = wid >> 2;         // 2 waves along N
    const int m0   = blockIdx.y * 128;
    const int n0   = blockIdx.x * 128;

    v8f acc[2][4];
#pragma unroll
    for (int mt = 0; mt < 2; ++mt)
#pragma unroll
        for (int nt = 0; nt < 4; ++nt)
            acc[mt][nt] = (v8f){0.f,0.f,0.f,0.f,0.f,0.f,0.f,0.f};

    const int arow = tid >> 1;
    const int acol = (tid & 1) * 16;

    for (int k0 = 0; k0 < K; k0 += 32) {
        // ---- stage A (bf16 -> bf16) ----
        const ushort_t* ap = A + (size_t)(m0 + arow) * lda + k0 + acol;
        *(uint4*)&sA[arow][acol]     = *(const uint4*)ap;
        *(uint4*)&sA[arow][acol + 8] = *(const uint4*)(ap + 8);
        // ---- stage B (f32 -> bf16 convert on the fly) ----
        const float* bp = B + (size_t)(n0 + arow) * ldb + k0 + acol;
#pragma unroll
        for (int i = 0; i < 16; i += 2) {
            unsigned lo = f2b(bp[i]);
            unsigned hi = f2b(bp[i + 1]);
            *(unsigned*)&sB[arow][acol + i] = lo | (hi << 16);
        }
        if (k0 + 32 < K) {             // hint next tiles (global_prefetch_b8)
            __builtin_prefetch(ap + 32, 0, 0);
            __builtin_prefetch(bp + 32, 0, 0);
        }
        __syncthreads();

        v16bf af[2], bfv[4];
#pragma unroll
        for (int mt = 0; mt < 2; ++mt)
            af[mt] = frag_ld(&sA[wm * 32 + mt * 16][0], 40, lane, 0);
#pragma unroll
        for (int nt = 0; nt < 4; ++nt)
            bfv[nt] = frag_ld(&sB[wn * 64 + nt * 16][0], 40, lane, 0);
#pragma unroll
        for (int mt = 0; mt < 2; ++mt)
#pragma unroll
            for (int nt = 0; nt < 4; ++nt)
                acc[mt][nt] = wmma_bf16(af[mt], bfv[nt], acc[mt][nt]);
        __syncthreads();
    }

    // Epilogue: C layout -> lane&15 = N col, lane-half picks rows 0-7/8-15.
    const int nl = lane & 15;
    const int mb = (lane >> 4) * 8;
#pragma unroll
    for (int mt = 0; mt < 2; ++mt) {
#pragma unroll
        for (int nt = 0; nt < 4; ++nt) {
            const int n = n0 + wn * 64 + nt * 16 + nl;
#pragma unroll
            for (int r = 0; r < 8; ++r) {
                const int m = m0 + wm * 32 + mt * 16 + mb + r;
                float v = acc[mt][nt][r];
                if (RESID) v += R[(size_t)m * ldr + n];
                C[(size_t)m * ldc + n] = v;
            }
        }
    }
}

// ---------------------------------------------------------------------
// RMS norms
// ---------------------------------------------------------------------
__global__ __launch_bounds__(256)
void rmsnorm_l2(const float* __restrict__ x, ushort_t* __restrict__ xn)
{
    const int row = blockIdx.x;
    const float* xr = x + (size_t)row * HID;
    __shared__ float red[8];
    float s = 0.f;
    for (int c = threadIdx.x; c < HID; c += 256) { float v = xr[c]; s += v * v; }
#pragma unroll
    for (int o = 16; o; o >>= 1) s += __shfl_down(s, o, 32);
    if ((threadIdx.x & 31) == 0) red[threadIdx.x >> 5] = s;
    __syncthreads();
    if (threadIdx.x == 0) {
        float t = 0.f;
#pragma unroll
        for (int i = 0; i < 8; ++i) t += red[i];
        red[0] = rsqrtf(t);            // reference: rsqrt(sum(x*x)), no eps
    }
    __syncthreads();
    const float rs = red[0];
    for (int c = threadIdx.x; c < HID; c += 256)
        xn[(size_t)row * HID + c] = f2b(xr[c] * rs);
}

__global__ __launch_bounds__(256)
void rmsnorm_final(float* __restrict__ x, const float* __restrict__ nw,
                   ushort_t* __restrict__ hn_bf)
{
    const int row = blockIdx.x;
    float* xr = x + (size_t)row * HID;
    __shared__ float red[8];
    float s = 0.f;
    for (int c = threadIdx.x; c < HID; c += 256) { float v = xr[c]; s += v * v; }
#pragma unroll
    for (int o = 16; o; o >>= 1) s += __shfl_down(s, o, 32);
    if ((threadIdx.x & 31) == 0) red[threadIdx.x >> 5] = s;
    __syncthreads();
    if (threadIdx.x == 0) {
        float t = 0.f;
#pragma unroll
        for (int i = 0; i < 8; ++i) t += red[i];
        red[0] = rsqrtf(t * (1.0f / HID) + 1e-6f);
    }
    __syncthreads();
    const float rs = red[0];
    for (int c = threadIdx.x; c < HID; c += 256) {
        float v = xr[c] * rs * nw[c];
        xr[c] = v;                                   // hn written in place
        hn_bf[(size_t)row * HID + c] = f2b(v);
    }
}

// ---------------------------------------------------------------------
// RoPE on Q; K/V concat with cache (K -> [kv,64] bf16, V -> [64,kv] bf16)
// ---------------------------------------------------------------------
__device__ __forceinline__ float rope_val(const float* base, int d, float p) {
    const int i = d & 31;
    const float fr = p * __powf(10000.0f, -(float)(2 * i) / 64.0f);
    const float c = __cosf(fr), sn = __sinf(fr);
    const float v = base[d];
    return (d < 32) ? (v * c - base[d + 32] * sn)
                    : (v * c + base[d - 32] * sn);
}

__global__ __launch_bounds__(64)
void rope_q(const float* __restrict__ qkv, ushort_t* __restrict__ q_bf)
{
    const int t = blockIdx.x, h = blockIdx.y, d = threadIdx.x;
    const float p = (float)(HIST + t);
    const float* base = qkv + (size_t)t * QKVW + h * HD;
    q_bf[((size_t)h * TQ + t) * HD + d] = f2b(rope_val(base, d, p));
}

__global__ __launch_bounds__(64)
void kv_concat(const float* __restrict__ qkv,
               const float* __restrict__ kc,   // [2][64][HIST] (layer slice)
               const float* __restrict__ vc,   // [2][HIST][64]
               ushort_t* __restrict__ k_all,   // [2][KVSEQ][64]
               ushort_t* __restrict__ v_allT)  // [2][64][KVSEQ]
{
    const int j = blockIdx.x, kvh = blockIdx.y, d = threadIdx.x;
    float kv, vv;
    if (j < HIST) {
        kv = kc[((size_t)kvh * HD + d) * HIST + j];
        vv = vc[((size_t)kvh * HIST + j) * HD + d];
    } else {
        const int t = j - HIST;
        const float* kb = qkv + (size_t)t * QKVW + QOUT + kvh * HD;
        kv = rope_val(kb, d, (float)j);
        vv = qkv[(size_t)t * QKVW + QOUT + 128 + kvh * HD + d];
    }
    k_all[((size_t)kvh * KVSEQ + j) * HD + d]   = f2b(kv);
    v_allT[((size_t)kvh * HD + d) * KVSEQ + j]  = f2b(vv);
}

// ---------------------------------------------------------------------
// Flash attention: grid (16 q-blocks of 64, 16 heads), 128 threads.
// Each wave owns 16 q rows. Streams kv in tiles of 64 with online softmax.
// Mask per reference: s += (j > t) ? -128*mask0 : 0  (NOT -inf, no scale).
// ---------------------------------------------------------------------
__global__ __launch_bounds__(128)
void flash_attn(const ushort_t* __restrict__ q_bf,
                const ushort_t* __restrict__ k_all,
                const ushort_t* __restrict__ v_allT,
                ushort_t* __restrict__ o_bf,
                const float* __restrict__ maskp)
{
    __shared__ ushort_t sQ[64][72];
    __shared__ ushort_t sK[64][72];
    __shared__ ushort_t sV[64][72];
    __shared__ float    sS[4][16][64];
    __shared__ ushort_t sP[4][16][72];
    __shared__ float    sM[4][16], sL[4][16], sFac[4][16];

    const int tid  = threadIdx.x;
    const int lane = tid & 31;
    const int w    = tid >> 5;
    const int qb   = blockIdx.x;
    const int h    = blockIdx.y;
    const int kvh  = h >> 3;
    const float nm = -128.0f * maskp[0];

    // Q tile (64 x 64) once.
    for (int i = tid; i < 64 * 8; i += 128) {
        const int r = i >> 3, c8 = (i & 7) * 8;
        *(uint4*)&sQ[r][c8] =
            *(const uint4*)(q_bf + ((size_t)h * TQ + qb * 64 + r) * HD + c8);
    }
    if (tid < 64) { sM[tid >> 4][tid & 15] = -3.0e38f; sL[tid >> 4][tid & 15] = 0.f; }

    v8f o_acc[4];
#pragma unroll
    for (int nt = 0; nt < 4; ++nt)
        o_acc[nt] = (v8f){0.f,0.f,0.f,0.f,0.f,0.f,0.f,0.f};

    const int nl = lane & 15;
    const int mb = (lane >> 4) * 8;

    for (int jt = 0; jt < KVSEQ / 64; ++jt) {
        __syncthreads();
        // stage K (kv-rows x dim) and V^T (dim-rows x kv-cols)
        for (int i = tid; i < 64 * 8; i += 128) {
            const int r = i >> 3, c8 = (i & 7) * 8;
            *(uint4*)&sK[r][c8] = *(const uint4*)
                (k_all + ((size_t)kvh * KVSEQ + jt * 64 + r) * HD + c8);
            *(uint4*)&sV[r][c8] = *(const uint4*)
                (v_allT + ((size_t)kvh * HD + r) * KVSEQ + jt * 64 + c8);
        }
        __syncthreads();

        // S = Q * K^T  (per wave, 16x64)
#pragma unroll
        for (int nt = 0; nt < 4; ++nt) {
            v8f acc = (v8f){0.f,0.f,0.f,0.f,0.f,0.f,0.f,0.f};
#pragma unroll
            for (int ks = 0; ks < 2; ++ks) {
                v16bf a = frag_ld(&sQ[w * 16][0], 72, lane, ks * 32);
                v16bf b = frag_ld(&sK[nt * 16][0], 72, lane, ks * 32);
                acc = wmma_bf16(a, b, acc);
            }
            const int n = nt * 16 + nl;
            const int j = jt * 64 + n;
#pragma unroll
            for (int r = 0; r < 8; ++r) {
                const int t = qb * 64 + w * 16 + mb + r;
                sS[w][mb + r][n] = acc[r] + ((j > t) ? nm : 0.f);
            }
        }
        __syncthreads();

        // online softmax: lanes 0..15 each own one row
        if (lane < 16) {
            float* row = sS[w][lane];
            const float mo = sM[w][lane], lo = sL[w][lane];
            float tm = row[0];
#pragma unroll 8
            for (int j2 = 1; j2 < 64; ++j2) tm = fmaxf(tm, row[j2]);
            const float mn  = fmaxf(mo, tm);
            const float fac = __expf(mo - mn);
            float ssum = 0.f;
#pragma unroll 8
            for (int j2 = 0; j2 < 64; ++j2) {
                const float pv = __expf(row[j2] - mn);
                ssum += pv;
                sP[w][lane][j2] = f2b(pv);
            }
            sM[w][lane] = mn;
            sL[w][lane] = lo * fac + ssum;
            sFac[w][lane] = fac;
        }
        __syncthreads();

        // O = O*fac + P * V
        float fr[8];
#pragma unroll
        for (int r = 0; r < 8; ++r) fr[r] = sFac[w][mb + r];
#pragma unroll
        for (int nt = 0; nt < 4; ++nt) {
#pragma unroll
            for (int r = 0; r < 8; ++r) o_acc[nt][r] *= fr[r];
#pragma unroll
            for (int ks = 0; ks < 2; ++ks) {
                v16bf a = frag_ld(&sP[w][0][0], 72, lane, ks * 32);
                v16bf b = frag_ld(&sV[nt * 16][0], 72, lane, ks * 32);
                o_acc[nt] = wmma_bf16(a, b, o_acc[nt]);
            }
        }
    }
    __syncthreads();

    float lr[8];
#pragma unroll
    for (int r = 0; r < 8; ++r) lr[r] = sL[w][mb + r];
#pragma unroll
    for (int nt = 0; nt < 4; ++nt) {
        const int d = nt * 16 + nl;
#pragma unroll
        for (int r = 0; r < 8; ++r) {
            const int t = qb * 64 + w * 16 + mb + r;
            o_bf[(size_t)t * HID + h * HD + d] = f2b(o_acc[nt][r] / lr[r]);
        }
    }
}

// ---------------------------------------------------------------------
// silu(gate) * up  -> bf16
// ---------------------------------------------------------------------
__global__ __launch_bounds__(256)
void silu_mul(const float* __restrict__ gu, ushort_t* __restrict__ act)
{
    const int i = blockIdx.x * 256 + threadIdx.x;   // over TQ*DFF
    const int t = i >> 12, j = i & (DFF - 1);
    const float g = gu[(size_t)t * 2 * DFF + j];
    const float u = gu[(size_t)t * 2 * DFF + DFF + j];
    act[i] = f2b(g / (1.0f + __expf(-g)) * u);
}

// =====================================================================
// Host orchestration
// =====================================================================
extern "C" void kernel_launch(void* const* d_in, const int* in_sizes, int n_in,
                              void* d_out, int out_size, void* d_ws, size_t ws_size,
                              hipStream_t stream)
{
    (void)in_sizes; (void)n_in; (void)out_size; (void)ws_size;

    const float* hs     = (const float*)d_in[0];   // (1, 1024, 1024)
    const float* fe     = (const float*)d_in[1];   // (1, 512, 1024)
    const float* qkv_w  = (const float*)d_in[2];   // (12, 1280, 1024)
    const float* o_w    = (const float*)d_in[3];   // (12, 1024, 1024)
    const float* gu_w   = (const float*)d_in[4];   // (12, 8192, 1024)
    const float* dn_w   = (const float*)d_in[5];   // (12, 1024, 4096)
    const float* nw     = (const float*)d_in[6];   // (1024,)
    const float* fsq_w  = (const float*)d_in[7];   // (1024, 1024)
    const float* kc     = (const float*)d_in[8];   // (12, 2, 1, 64, 256)
    const float* vc     = (const float*)d_in[9];   // (12, 2, 1, 256, 64)
    const float* maskp  = (const float*)d_in[10];  // (1,)

    // -------- workspace carve (all 256B aligned; ~58 MB total) --------
    char* w = (char*)d_ws;
    size_t off = 0;
    auto take = [&](size_t bytes) -> char* {
        char* p = w + off;
        off += (bytes + 255) & ~(size_t)255;
        return p;
    };
    float*    x      = (float*)   take((size_t)TQ * HID * 4);        // 4 MB
    ushort_t* xn     = (ushort_t*)take((size_t)TQ * HID * 2);        // 2 MB
    float*    qkvb   = (float*)   take((size_t)TQ * QKVW * 4);       // 5 MB
    ushort_t* q_bf   = (ushort_t*)take((size_t)NHEAD * TQ * HD * 2); // 2 MB
    ushort_t* k_all  = (ushort_t*)take((size_t)2 * KVSEQ * HD * 2);
    ushort_t* v_allT = (ushort_t*)take((size_t)2 * HD * KVSEQ * 2);
    ushort_t* o_bf   = (ushort_t*)take((size_t)TQ * HID * 2);        // 2 MB
    float*    gu     = (float*)   take((size_t)TQ * 2 * DFF * 4);    // 32 MB
    ushort_t* act    = (ushort_t*)take((size_t)TQ * DFF * 2);        // 8 MB
    ushort_t* hn_bf  = (ushort_t*)take((size_t)TQ * HID * 2);        // 2 MB

    hipMemcpyAsync(x, hs, (size_t)TQ * HID * 4, hipMemcpyDeviceToDevice, stream);

    auto run_layer = [&](int l) {
        // attention block
        rmsnorm_l2<<<TQ, 256, 0, stream>>>(x, xn);
        gemm_nt_f32w<false><<<dim3(QKVW / 128, TQ / 128), 256, 0, stream>>>(
            xn, HID, qkv_w + (size_t)l * QKVW * HID, HID,
            qkvb, QKVW, nullptr, 0, HID);
        rope_q<<<dim3(TQ, NHEAD), 64, 0, stream>>>(qkvb, q_bf);
        kv_concat<<<dim3(KVSEQ, 2), 64, 0, stream>>>(
            qkvb, kc + (size_t)l * 2 * HD * HIST,
            vc + (size_t)l * 2 * HIST * HD, k_all, v_allT);
        flash_attn<<<dim3(TQ / 64, NHEAD), 128, 0, stream>>>(
            q_bf, k_all, v_allT, o_bf, maskp);
        gemm_nt_f32w<true><<<dim3(HID / 128, TQ / 128), 256, 0, stream>>>(
            o_bf, HID, o_w + (size_t)l * HID * HID, HID,
            x, HID, x, HID, HID);
        // MLP block
        rmsnorm_l2<<<TQ, 256, 0, stream>>>(x, xn);
        gemm_nt_f32w<false><<<dim3(2 * DFF / 128, TQ / 128), 256, 0, stream>>>(
            xn, HID, gu_w + (size_t)l * 2 * DFF * HID, HID,
            gu, 2 * DFF, nullptr, 0, HID);
        silu_mul<<<(TQ * DFF) / 256, 256, 0, stream>>>(gu, act);
        gemm_nt_f32w<true><<<dim3(HID / 128, TQ / 128), 256, 0, stream>>>(
            act, DFF, dn_w + (size_t)l * HID * DFF, DFF,
            x, HID, x, HID, DFF);
    };

    for (int l = 0; l < 6; ++l) run_layer(l);

    // mid-stack: final-style norm, then fsq on rows [512:1024) + feat_embed
    rmsnorm_final<<<TQ, 256, 0, stream>>>(x, nw, hn_bf);
    gemm_nt_f32w<true><<<dim3(HID / 128, 512 / 128), 256, 0, stream>>>(
        hn_bf + (size_t)512 * HID, HID, fsq_w, HID,
        x + (size_t)512 * HID, HID, fe, HID, HID);

    for (int l = 6; l < 12; ++l) run_layer(l);

    hipMemcpyAsync(d_out, x, (size_t)TQ * HID * 4,
                   hipMemcpyDeviceToDevice, stream);
}
